// MultiHeadSelfAttention_40836549050591
// MI455X (gfx1250) — matrix operand order repaired and checked by reading, compiled, not run on hardware
//
#include <hip/hip_runtime.h>

typedef _Float16 f16;
typedef __attribute__((ext_vector_type(16))) _Float16 v16h;
typedef __attribute__((ext_vector_type(8)))  _Float16 v8h;
typedef __attribute__((ext_vector_type(8)))  float    v8f;

#define BATCH   4
#define SEQ     2048
#define DMODEL  1024
#define NHEADS  16
#define DK      64
// ln(10000)/32  (RoPE: inv_freq[p] = 10000^(-p/32), dk=64)
#define ROPE_C  0.28782313662425575f

union AFrag { v16h v; v8h h[2]; };

static __device__ __forceinline__ v8f vzero8() {
    v8f z;
#pragma unroll
    for (int r = 0; r < 8; ++r) z[r] = 0.0f;
    return z;
}

// ---------------------------------------------------------------------------
// fp32 -> f16 cast
// ---------------------------------------------------------------------------
__global__ void cast_f32_f16(const float* __restrict__ src, f16* __restrict__ dst, int n) {
    int i = blockIdx.x * blockDim.x + threadIdx.x;
    if (i < n) dst[i] = (f16)src[i];
}

// ---------------------------------------------------------------------------
// WMMA GEMM:  C[M,N] = A[M,K] * Bw[N,K]^T      (A,Bw row-major f16)
// One wave per 64x64 tile. 16 v_wmma_f32_16x16x32_f16 per K-step of 32.
// EPI==0 : write fp32 to out[M,N]
// EPI==1 : fused QKV epilogue (N=3072): RoPE(Q,K), Q*=1/8,
//          scatter Q->[bh][s][64], K->[bh][s][64], V->[bh][64][s] (f16)
// ---------------------------------------------------------------------------
template <int EPI>
__global__ __launch_bounds__(32)
void gemm_wmma(const f16* __restrict__ A, const f16* __restrict__ Bw,
               int M, int N, int K,
               float* __restrict__ out,
               f16* __restrict__ Qh, f16* __restrict__ Kh, f16* __restrict__ Vt,
               const int* __restrict__ pos) {
    const int lane = threadIdx.x & 31;
    const int ln = lane & 15;
    const int lh = lane >> 4;
    const int m0 = blockIdx.x * 64;
    const int n0 = blockIdx.y * 64;

    v8f acc[4][4];
#pragma unroll
    for (int i = 0; i < 4; ++i)
#pragma unroll
        for (int j = 0; j < 4; ++j) acc[i][j] = vzero8();

    const f16* aLane0 = A  + (long)(m0 + ln) * K + 8 * lh;
    const f16* bLane0 = Bw + (long)(n0 + ln) * K + 16 * lh;

    for (int k0 = 0; k0 < K; k0 += 32) {
        AFrag af[4];
#pragma unroll
        for (int mt = 0; mt < 4; ++mt) {
            const f16* p = A + (long)(m0 + mt * 16 + ln) * K + k0 + 16 * lh;
            af[mt].h[0] = *(const v8h*)p;
            af[mt].h[1] = *(const v8h*)(p + 8);
        }
        v16h bf[4];
#pragma unroll
        for (int nt = 0; nt < 4; ++nt) {
            const f16* p = Bw + (long)(n0 + nt * 16 + ln) * K + k0 + 16 * lh;
            bf[nt] = *(const v16h*)p;
        }
        if (k0 + 32 < K) {  // gfx1250 global_prefetch_b8 for next K-slab
            __builtin_prefetch(aLane0 + k0 + 32, 0, 1);
            __builtin_prefetch(bLane0 + k0 + 32, 0, 1);
        }
#pragma unroll
        for (int mt = 0; mt < 4; ++mt)
#pragma unroll
            for (int nt = 0; nt < 4; ++nt)
                acc[mt][nt] = __builtin_amdgcn_wmma_f32_16x16x32_f16(
                    false, af[mt].v, false, bf[nt], (short)0, acc[mt][nt],
                    false, false);
    }

    if (EPI == 0) {
#pragma unroll
        for (int mt = 0; mt < 4; ++mt)
#pragma unroll
            for (int nt = 0; nt < 4; ++nt) {
                const int col = n0 + nt * 16 + ln;
#pragma unroll
                for (int r = 0; r < 8; ++r) {
                    const int row = m0 + mt * 16 + r + 8 * lh;
                    out[(long)row * N + col] = acc[mt][nt][r];
                }
            }
    } else {
        const int region = n0 >> 10;           // 0=Q 1=K 2=V
        const int head   = (n0 & 1023) >> 6;   // 64-wide heads
#pragma unroll
        for (int mt = 0; mt < 4; ++mt)
#pragma unroll
            for (int nt = 0; nt < 4; ++nt) {
                const int d   = nt * 16 + ln;      // head-dim index 0..63
                const int p2  = d >> 1;
                const float invf = __expf(-(float)p2 * ROPE_C);
                const float sgn  = (d & 1) ? 1.0f : -1.0f;
#pragma unroll
                for (int r = 0; r < 8; ++r) {
                    const int i = m0 + mt * 16 + r + 8 * lh;   // flat (b,s)
                    const int b = i >> 11;                     // /SEQ
                    const int s = i & (SEQ - 1);
                    float v = acc[mt][nt][r];
                    if (region <= 1) {                         // RoPE on Q,K
                        const float prt = __shfl_xor(v, 1, 32);
                        const float ang = (float)pos[s] * invf;
                        float sv, cv;
                        __sincosf(ang, &sv, &cv);
                        v = v * cv + sgn * prt * sv;
                    }
                    const long bh = (long)b * NHEADS + head;
                    if (region == 0) {
                        v *= 0.125f;                           // 1/sqrt(dk)
                        Qh[(bh * SEQ + s) * DK + d] = (f16)v;
                    } else if (region == 1) {
                        Kh[(bh * SEQ + s) * DK + d] = (f16)v;
                    } else {
                        Vt[(bh * DK + d) * SEQ + s] = (f16)v;  // transposed V
                    }
                }
            }
    }
}

// ---------------------------------------------------------------------------
// Causal flash attention: one wave per (b,h, 32-row q-block = 2 x 16-row
// sub-tiles sharing every K/V fragment). Q.K^T, row-sums (P x ones) and P.V
// all through v_wmma. Running max via stage-outer ds_bpermute reductions so
// each stage's 16 shuffles share one s_wait_dscnt.
// ---------------------------------------------------------------------------
__global__ __launch_bounds__(32)
void attn_wmma(const f16* __restrict__ Qh, const f16* __restrict__ Kh,
               const f16* __restrict__ Vt, f16* __restrict__ Oh) {
    const int lane = threadIdx.x & 31;
    const int ln = lane & 15;
    const int lh = lane >> 4;
    const int qb = blockIdx.x;            // 0..63 : rows qb*32 .. qb*32+31
    const int bh = blockIdx.y;            // 0..63
    const int b  = bh >> 4;
    const int h  = bh & 15;
    const long qkBase = (long)bh * SEQ * DK;
    const long vtBase = (long)bh * DK * SEQ;

    __shared__ __align__(16) f16 Pl[2][16 * 32];

    // all-ones B fragment for WMMA row sums
    v16h ones;
#pragma unroll
    for (int e = 0; e < 16; ++e) ones[e] = (f16)1.0f;

    AFrag qf[2][2];
#pragma unroll
    for (int mq = 0; mq < 2; ++mq)
#pragma unroll
        for (int c = 0; c < 2; ++c) {
            const f16* p = Qh + qkBase + (long)(qb * 32 + mq * 16 + ln) * DK +
                           c * 32 + 16 * lh;
            qf[mq][c].h[0] = *(const v8h*)p;
            qf[mq][c].h[1] = *(const v8h*)(p + 8);
        }

    v8f o[2][4];
    float mrun[2][8], lrun[2][8];
#pragma unroll
    for (int mq = 0; mq < 2; ++mq) {
#pragma unroll
        for (int t = 0; t < 4; ++t) o[mq][t] = vzero8();
#pragma unroll
        for (int r = 0; r < 8; ++r) { mrun[mq][r] = -1e30f; lrun[mq][r] = 0.0f; }
    }

    const int nChunks = qb + 1;          // 32 keys per chunk, causal
    for (int kc = 0; kc < nChunks; ++kc) {
        const int key0 = kc * 32;
        // ---- K fragments (shared by both q sub-tiles) ----
        v16h kf[2][2];
#pragma unroll
        for (int t = 0; t < 2; ++t) {
            const f16* kp = Kh + qkBase + (long)(key0 + t * 16 + ln) * DK + 16 * lh;
            kf[t][0] = *(const v16h*)kp;
            kf[t][1] = *(const v16h*)(kp + 32);
        }
        // ---- scores ----
        v8f st[2][2];
#pragma unroll
        for (int mq = 0; mq < 2; ++mq)
#pragma unroll
            for (int t = 0; t < 2; ++t) {
                v8f z = vzero8();
                z = __builtin_amdgcn_wmma_f32_16x16x32_f16(
                    false, qf[mq][0].v, false, kf[t][0], (short)0, z, false, false);
                z = __builtin_amdgcn_wmma_f32_16x16x32_f16(
                    false, qf[mq][1].v, false, kf[t][1], (short)0, z, false, false);
                st[mq][t] = z;
            }
        // ---- causal mask + lane-local max ----
        float mx[2][8];
#pragma unroll
        for (int mq = 0; mq < 2; ++mq)
#pragma unroll
            for (int r = 0; r < 8; ++r) {
                const int srow = qb * 32 + mq * 16 + r + 8 * lh;
                float m = mrun[mq][r];
#pragma unroll
                for (int t = 0; t < 2; ++t) {
                    const int kcol = key0 + t * 16 + ln;
                    float v = st[mq][t][r];
                    v = (kcol <= srow) ? v : -1e30f;
                    st[mq][t][r] = v;
                    m = fmaxf(m, v);
                }
                mx[mq][r] = m;
            }
        // ---- stage-outer 16-lane max reduction (batched bpermutes) ----
#pragma unroll
        for (int off = 1; off < 16; off <<= 1) {
            float tmp[2][8];
#pragma unroll
            for (int mq = 0; mq < 2; ++mq)
#pragma unroll
                for (int r = 0; r < 8; ++r)
                    tmp[mq][r] = __shfl_xor(mx[mq][r], off, 32);
#pragma unroll
            for (int mq = 0; mq < 2; ++mq)
#pragma unroll
                for (int r = 0; r < 8; ++r)
                    mx[mq][r] = fmaxf(mx[mq][r], tmp[mq][r]);
        }
        // ---- exponentials + rescale O ----
        float scl[2][8];
#pragma unroll
        for (int mq = 0; mq < 2; ++mq)
#pragma unroll
            for (int r = 0; r < 8; ++r) {
                scl[mq][r] = __expf(mrun[mq][r] - mx[mq][r]);
                mrun[mq][r] = mx[mq][r];
#pragma unroll
                for (int t = 0; t < 2; ++t)
                    st[mq][t][r] = __expf(st[mq][t][r] - mx[mq][r]);
            }
#pragma unroll
        for (int mq = 0; mq < 2; ++mq)
#pragma unroll
            for (int t4 = 0; t4 < 4; ++t4)
#pragma unroll
                for (int r = 0; r < 8; ++r) o[mq][t4][r] *= scl[mq][r];

        // ---- P: C-layout -> A-layout through LDS ----
        __syncthreads();
#pragma unroll
        for (int mq = 0; mq < 2; ++mq)
#pragma unroll
            for (int t = 0; t < 2; ++t)
#pragma unroll
                for (int r = 0; r < 8; ++r)
                    Pl[mq][(r + 8 * lh) * 32 + t * 16 + ln] = (f16)st[mq][t][r];
        __syncthreads();
        AFrag pf[2];
#pragma unroll
        for (int mq = 0; mq < 2; ++mq) {
            const f16* p = Pl[mq] + ln * 32 + 16 * lh;
            pf[mq].h[0] = *(const v8h*)p;
            pf[mq].h[1] = *(const v8h*)(p + 8);
        }
        // ---- row sums via WMMA: (P x ones) -> every lane holds its row sum
        //      in the same (VGPR r, half) slot as lrun ----
#pragma unroll
        for (int mq = 0; mq < 2; ++mq) {
            v8f ls = vzero8();
            ls = __builtin_amdgcn_wmma_f32_16x16x32_f16(
                false, pf[mq].v, false, ones, (short)0, ls, false, false);
#pragma unroll
            for (int r = 0; r < 8; ++r)
                lrun[mq][r] = lrun[mq][r] * scl[mq][r] + ls[r];
        }
        // ---- O += P.V ----
#pragma unroll
        for (int t4 = 0; t4 < 4; ++t4) {
            v16h vf = *(const v16h*)(Vt + vtBase + (long)(t4 * 16 + ln) * SEQ +
                                     key0 + 16 * lh);
#pragma unroll
            for (int mq = 0; mq < 2; ++mq)
                o[mq][t4] = __builtin_amdgcn_wmma_f32_16x16x32_f16(
                    false, pf[mq].v, false, vf, (short)0, o[mq][t4], false, false);
        }
    }

    // ---- normalize + write O -> [b][s][h*64+d] f16 ----
#pragma unroll
    for (int mq = 0; mq < 2; ++mq)
#pragma unroll
        for (int t4 = 0; t4 < 4; ++t4)
#pragma unroll
            for (int r = 0; r < 8; ++r) {
                const int srow = qb * 32 + mq * 16 + r + 8 * lh;
                const int d = t4 * 16 + ln;
                const float v = o[mq][t4][r] / lrun[mq][r];
                Oh[((long)b * SEQ + srow) * DMODEL + h * DK + d] = (f16)v;
            }
}

// ---------------------------------------------------------------------------
extern "C" void kernel_launch(void* const* d_in, const int* in_sizes, int n_in,
                              void* d_out, int out_size, void* d_ws, size_t ws_size,
                              hipStream_t stream) {
    const float* x  = (const float*)d_in[0];
    const float* Wq = (const float*)d_in[1];
    const float* Wk = (const float*)d_in[2];
    const float* Wv = (const float*)d_in[3];
    const float* Wo = (const float*)d_in[4];
    const int*   tp = (const int*)d_in[5];
    float* out = (float*)d_out;

    const int M  = BATCH * SEQ;          // 8192
    const int nX = M * DMODEL;           // 8M
    const int nW = DMODEL * DMODEL;      // 1M

    char* ws = (char*)d_ws;
    size_t off = 0;
    f16* xh    = (f16*)(ws + off); off += (size_t)nX * 2;        // 16 MiB
    f16* Wqkvh = (f16*)(ws + off); off += (size_t)3 * nW * 2;    //  6 MiB
    f16* Woh   = (f16*)(ws + off); off += (size_t)nW * 2;        //  2 MiB
    f16* Qh    = (f16*)(ws + off); off += (size_t)nX * 2;        // 16 MiB
    f16* Kh    = (f16*)(ws + off); off += (size_t)nX * 2;        // 16 MiB
    f16* Vt    = (f16*)(ws + off); off += (size_t)nX * 2;        // 16 MiB
    f16* Oh    = (f16*)(ws + off); off += (size_t)nX * 2;        // 16 MiB

    // casts (weights stay row-major: [N][K] is the native WMMA B layout)
    cast_f32_f16<<<(nX + 255) / 256, 256, 0, stream>>>(x, xh, nX);
    cast_f32_f16<<<(nW + 255) / 256, 256, 0, stream>>>(Wq, Wqkvh, nW);
    cast_f32_f16<<<(nW + 255) / 256, 256, 0, stream>>>(Wk, Wqkvh + nW, nW);
    cast_f32_f16<<<(nW + 255) / 256, 256, 0, stream>>>(Wv, Wqkvh + 2 * nW, nW);
    cast_f32_f16<<<(nW + 255) / 256, 256, 0, stream>>>(Wo, Woh, nW);

    // fused QKV projection + RoPE + scatter
    dim3 g1(M / 64, (3 * DMODEL) / 64);
    gemm_wmma<1><<<g1, 32, 0, stream>>>(xh, Wqkvh, M, 3 * DMODEL, DMODEL,
                                        nullptr, Qh, Kh, Vt, tp);

    // causal flash attention (32 query rows per wave)
    dim3 g2(SEQ / 32, BATCH * NHEADS);
    attn_wmma<<<g2, 32, 0, stream>>>(Qh, Kh, Vt, Oh);

    // output projection -> fp32
    dim3 g3(M / 64, DMODEL / 64);
    gemm_wmma<0><<<g3, 32, 0, stream>>>(Oh, Woh, M, DMODEL, DMODEL,
                                        out, nullptr, nullptr, nullptr, nullptr);
}
